// RecurrentHierarchicalTransformer_1168231104799
// MI455X (gfx1250) — matrix operand order repaired and checked by reading, not compiled, not run on hardware
//
#include <hip/hip_runtime.h>

typedef __attribute__((ext_vector_type(16))) __bf16 v16bf;
typedef __attribute__((ext_vector_type(8)))  float  v8f;

#define B_    2
#define C_    64
#define T_    800
#define D_    128
#define H_    8
#define DH_   64
#define FFD_  512
#define L_    2
#define NC_   4
#define PL_   25
#define NW_   32
#define AHD_  512
#define NTOK  2049            // C*NW + 1
#define ROWS  (B_*NTOK)       // 4098
#define NPAD  2080            // key padding, multiple of 32
#define QT    129             // ceil(NTOK/16) query tiles per group
#define GRP   16              // pseudo (b,h) groups (see reshape analysis)
#define WPB   8               // waves (query tiles) per block in flash kernel
#define NCH   (NPAD/32)       // 65 key chunks of 32
#define SCALEF 0.125f         // DH^-0.5

// ---------------- WMMA helpers (CDNA5 gfx1250, wave32) ----------------

static __device__ __forceinline__ v8f wmma_bf16(v16bf a, v16bf b, v8f c) {
  // D(f32 16x16) = A(bf16 16x32) * B(bf16 32x16) + C
  return __builtin_amdgcn_wmma_f32_16x16x32_bf16(false, a, false, b, (short)0, c,
                                                 false, false);
}

// A-operand (16-bit 16x32): lane holds row M=lane%16; element e -> K =
// (e>=8)*16 + (lane>=16)*8 + (e&7): two contiguous 8-runs.
static __device__ __forceinline__ v16bf ldA_bf16(const __bf16* rowp, int k0, int hi) {
  const __bf16* p = rowp + k0 + hi * 8;
  v16bf r;
#pragma unroll
  for (int e = 0; e < 8; ++e) { r[e] = p[e]; r[e + 8] = p[e + 16]; }
  return r;
}
static __device__ __forceinline__ v16bf ldA_f32(const float* rowp, int k0, int hi) {
  const float* p = rowp + k0 + hi * 8;
  v16bf r;
#pragma unroll
  for (int e = 0; e < 8; ++e) { r[e] = (__bf16)p[e]; r[e + 8] = (__bf16)p[e + 16]; }
  return r;
}
// B-operand (16-bit 32x16): lane holds col N=lane%16; element e -> K =
// (e>=8)*16 + (lane>=16)*8 + (e&7): two contiguous 8-runs.
static __device__ __forceinline__ v16bf ldB_bf16(const __bf16* rowp, int k0, int hi) {
  const __bf16* p = rowp + k0 + hi * 8;
  v16bf r;
#pragma unroll
  for (int e = 0; e < 8; ++e) { r[e] = p[e]; r[e + 8] = p[e + 16]; }
  return r;
}

// Async global->LDS copy of 16 bytes per lane (CDNA5 ASYNCcnt path).
static __device__ __forceinline__ void async_copy16(unsigned lds_off, const void* gaddr) {
  asm volatile("global_load_async_to_lds_b128 %0, %1, off"
               :
               : "v"(lds_off), "v"(gaddr)
               : "memory");
}
static __device__ __forceinline__ void wait_asynccnt0() {
  asm volatile("s_wait_asynccnt 0x0" ::: "memory");
}
static __device__ __forceinline__ void wait_dscnt0() {
  asm volatile("s_wait_dscnt 0x0" ::: "memory");
}

// ---------------- Embedding: conv patches + sinusoidal pos embs ----------------

__global__ void embed_kernel(const float* __restrict__ x_eeg,
                             const float* __restrict__ est,
                             const int* __restrict__ cvi,
                             const float* __restrict__ conv_w,
                             const float* __restrict__ conv_b,
                             const float* __restrict__ cls_token,
                             const float* __restrict__ cls_pos,
                             float* __restrict__ x, float* __restrict__ r_t,
                             float* __restrict__ r_c) {
  int tok = blockIdx.x % NTOK;
  int b = blockIdx.x / NTOK;
  int d = threadIdx.x;  // 0..127
  size_t o = ((size_t)b * NTOK + tok) * D_ + d;
  if (tok == 0) {
    x[o] = cls_token[d];
    r_t[o] = cls_pos[d];
    r_c[o] = cls_pos[d];
    return;
  }
  int c = (tok - 1) >> 5;
  int w = (tok - 1) & 31;
  const float* xp = x_eeg + ((size_t)b * C_ + c) * T_ + w * PL_;
  float s = conv_b[d];
#pragma unroll
  for (int p = 0; p < PL_; ++p) s += xp[p] * conv_w[d * PL_ + p];
  x[o] = s;
  int i = d & 63;
  float inv = powf(10000.0f, -(float)i / 64.0f);
  float tp = floorf(est[b] / 0.1f) + (float)w;
  float at = tp * inv;
  float ac = (float)cvi[b * C_ + c] * inv;
  r_t[o] = (d < 64) ? sinf(at) : cosf(at);
  r_c[o] = (d < 64) ? sinf(ac) : cosf(ac);
}

// ---------------- Weight transpose + cast: f32 [K][N] -> bf16 [N][K] ----------------

__global__ void transpose_cast(const float* __restrict__ in, __bf16* __restrict__ out,
                               int K, int N) {
  int idx = blockIdx.x * 256 + threadIdx.x;
  if (idx >= K * N) return;
  int n = idx / K;
  int k = idx - n * K;
  out[idx] = (__bf16)in[(size_t)k * N + n];
}

// ---------------- LayerNorm over D=128 cols, one wave per row ----------------

__global__ void layernorm_k(const float* __restrict__ x, const float* __restrict__ g,
                            const float* __restrict__ bb, float* __restrict__ y,
                            int rows) {
  int wave = threadIdx.x >> 5, lane = threadIdx.x & 31;
  int row = blockIdx.x * 8 + wave;
  if (row >= rows) return;
  const float* xr = x + (size_t)row * D_;
  float v[4];
#pragma unroll
  for (int i = 0; i < 4; ++i) v[i] = xr[i * 32 + lane];
  float s = v[0] + v[1] + v[2] + v[3];
#pragma unroll
  for (int off = 1; off < 32; off <<= 1) s += __shfl_xor(s, off);
  float mean = s * (1.0f / 128.0f);
  float d2 = 0.f;
#pragma unroll
  for (int i = 0; i < 4; ++i) { float t = v[i] - mean; d2 += t * t; }
#pragma unroll
  for (int off = 1; off < 32; off <<= 1) d2 += __shfl_xor(d2, off);
  float inv = rsqrtf(d2 * (1.0f / 128.0f) + 1e-5f);
#pragma unroll
  for (int i = 0; i < 4; ++i) {
    int c = i * 32 + lane;
    y[(size_t)row * D_ + c] = (v[i] - mean) * inv * g[c] + bb[c];
  }
}

// ---------------- Generic bf16-WMMA GEMM: C[M][N] = A[M][K](f32) * Wt[N][K](bf16)
// flags: 1=+bias[col], 4=gelu(exact), 2=+resid[row][col]; applied in that order.

__global__ void gemm_wmma(const float* __restrict__ A, const __bf16* __restrict__ Wt,
                          const float* __restrict__ bias, const float* __restrict__ resid,
                          float* __restrict__ Cout, int M, int K, int N,
                          int totalTiles, int nt64, int flags) {
  int wave = threadIdx.x >> 5, lane = threadIdx.x & 31, hi = lane >> 4, lo = lane & 15;
  int widx = blockIdx.x * 8 + wave;
  if (widx >= totalTiles) return;
  int tm = widx / nt64, tn = widx - tm * nt64;
  int rowA = tm * 16 + lo;
  if (rowA >= M) rowA = M - 1;
  v8f zero8 = {0.f, 0.f, 0.f, 0.f, 0.f, 0.f, 0.f, 0.f};
  v8f acc[4] = {zero8, zero8, zero8, zero8};
  const float* Arow = A + (size_t)rowA * K;
  for (int k0 = 0; k0 < K; k0 += 32) {
    v16bf a = ldA_f32(Arow, k0, hi);
#pragma unroll
    for (int t = 0; t < 4; ++t) {
      int ncol = tn * 64 + t * 16 + lo;
      v16bf bw = ldB_bf16(Wt + (size_t)ncol * K, k0, hi);
      acc[t] = wmma_bf16(a, bw, acc[t]);
    }
  }
#pragma unroll
  for (int t = 0; t < 4; ++t) {
#pragma unroll
    for (int r = 0; r < 8; ++r) {
      int row = tm * 16 + r + 8 * hi;  // D layout: M = r + 8*(lane/16)
      if (row < M) {
        int col = tn * 64 + t * 16 + lo;
        float v = acc[t][r];
        if (flags & 1) v += bias[col];
        if (flags & 4) v = 0.5f * v * (1.0f + erff(v * 0.70710678f));
        if (flags & 2) v += resid[(size_t)row * N + col];
        Cout[(size_t)row * N + col] = v;
      }
    }
  }
}

// ---------------- Attention operand prep (replicates the reference's raw reshapes)
// Pseudo group g = b2*8 + h2, pseudo row i; flat 64-row index G = i*16 + g.

__global__ void attn_prep_k(const float* __restrict__ qkv, const float* __restrict__ ktf,
                            const float* __restrict__ kcf, const float* __restrict__ bte,
                            const float* __restrict__ btr, const float* __restrict__ bcr,
                            const float* __restrict__ bce, __bf16* __restrict__ qeb,
                            __bf16* __restrict__ qrb, __bf16* __restrict__ kbb,
                            __bf16* __restrict__ k2b, __bf16* __restrict__ vtb) {
  int d = threadIdx.x;   // 0..63
  int i = blockIdx.x;    // 0..NPAD-1
  int g = blockIdx.y;    // 0..15
  float qe = 0.f, qr = 0.f, kk = 0.f, k2 = 0.f, vv = 0.f;
  if (i < NTOK) {
    int G = i * 16 + g;                 // < 2*8*NTOK
    int bb = G / (8 * NTOK);
    int rem = G - bb * (8 * NTOK);
    int hh = rem / NTOK;
    int nn = rem - hh * NTOK;
    size_t qoff = ((size_t)(bb * NTOK + nn)) * 1536 + hh * 64 + d;
    float q = qkv[qoff];
    kk = qkv[qoff + 512];
    vv = qkv[qoff + 1024];
    size_t f = (size_t)G * 64 + d;      // kt/kc reshape is a flat reinterpret
    k2 = ktf[f] + kcf[f];               // Bt+Ct = q_r . (kt+kc)
    int hd = (g & 7) * 64 + d;          // bias indexed by pseudo head h2 = g&7
    qe = q + bte[hd] + bce[hd];
    qr = q + btr[hd] + bcr[hd];
  }
  size_t oi = ((size_t)g * NPAD + i) * 64 + d;
  qeb[oi] = (__bf16)qe;
  qrb[oi] = (__bf16)qr;
  kbb[oi] = (__bf16)kk;
  k2b[oi] = (__bf16)k2;
  vtb[((size_t)g * 64 + d) * NPAD + i] = (__bf16)vv;  // V stored transposed [64][NPAD]
}

// ---------------- Flash attention: 8 waves/block share async-staged K/V tiles --------
// Per 32-key chunk the block stages K (32x64), K2 (32x64) and V^T (64x32) bf16
// tiles into LDS via global_load_async_to_lds_b128 (double-buffered, ASYNCcnt),
// then every wave runs 12 WMMAs against the shared tiles.

__global__ __launch_bounds__(WPB * 32) void flash_attn_k(
    const __bf16* __restrict__ qe, const __bf16* __restrict__ qr,
    const __bf16* __restrict__ kb, const __bf16* __restrict__ k2b,
    const __bf16* __restrict__ vtb, float* __restrict__ o) {
  __shared__ __bf16 kT[2][32 * 64];
  __shared__ __bf16 k2T[2][32 * 64];
  __shared__ __bf16 vT[2][64 * 32];
  __shared__ float ldsP[WPB][16 * 32];

  int tid = threadIdx.x;
  int wave = tid >> 5, lane = tid & 31, hi = lane >> 4, lo = lane & 15;
  int qtile0 = blockIdx.x * WPB + wave;
  bool valid = qtile0 < QT;
  int qtile = valid ? qtile0 : QT - 1;  // clamp: all waves run (block barriers)
  int g = blockIdx.y;
  int b2 = g >> 3, h2 = g & 7;

  const __bf16* kB = kb + (size_t)g * NPAD * DH_;
  const __bf16* k2B = k2b + (size_t)g * NPAD * DH_;
  const __bf16* vB = vtb + (size_t)g * DH_ * NPAD;

  // Stage one 32-key chunk: 768 x 16B tasks over 256 threads (3 per thread).
  auto stage = [&](int kt, int buf) {
#pragma unroll
    for (int r = 0; r < 3; ++r) {
      int task = r * 256 + tid;
      int kbase = kt * 32;
      if (task < 256) {          // K tile: 32 rows x 128B
        int row = task >> 3, seg = task & 7;
        async_copy16((unsigned)(uintptr_t)&kT[buf][row * 64 + seg * 8],
                     kB + (size_t)(kbase + row) * 64 + seg * 8);
      } else if (task < 512) {   // K2 tile
        int t2 = task - 256;
        int row = t2 >> 3, seg = t2 & 7;
        async_copy16((unsigned)(uintptr_t)&k2T[buf][row * 64 + seg * 8],
                     k2B + (size_t)(kbase + row) * 64 + seg * 8);
      } else {                   // V^T tile: 64 rows x 64B
        int t2 = task - 512;
        int row = t2 >> 2, seg = t2 & 3;
        async_copy16((unsigned)(uintptr_t)&vT[buf][row * 32 + seg * 8],
                     vB + (size_t)row * NPAD + kbase + seg * 8);
      }
    }
  };

  const __bf16* qeB = qe + ((size_t)g * NPAD + qtile * 16) * DH_;
  const __bf16* qrB = qr + ((size_t)g * NPAD + qtile * 16) * DH_;
  v16bf qe0 = ldA_bf16(qeB + (size_t)lo * DH_, 0, hi);
  v16bf qe1 = ldA_bf16(qeB + (size_t)lo * DH_, 32, hi);
  v16bf qr0 = ldA_bf16(qrB + (size_t)lo * DH_, 0, hi);
  v16bf qr1 = ldA_bf16(qrB + (size_t)lo * DH_, 32, hi);

  v8f zero8 = {0.f, 0.f, 0.f, 0.f, 0.f, 0.f, 0.f, 0.f};
  v8f accO[4] = {zero8, zero8, zero8, zero8};
  float m[8], l[8];
#pragma unroll
  for (int r = 0; r < 8; ++r) { m[r] = -1e30f; l[r] = 0.f; }
  float* myP = ldsP[wave];

  stage(0, 0);
  for (int kt = 0; kt < NCH; ++kt) {
    int buf = kt & 1;
    int kbase = kt * 32;
    wait_asynccnt0();     // my async copies done
    __syncthreads();      // everyone's copies done; prev reads of this buf done
    if (kt + 1 < NCH) stage(kt + 1, buf ^ 1);

    v8f s0 = zero8, s1 = zero8;
    {
      v16bf b00 = ldB_bf16(&kT[buf][lo * 64], 0, hi);
      v16bf b01 = ldB_bf16(&kT[buf][lo * 64], 32, hi);
      v16bf c00 = ldB_bf16(&k2T[buf][lo * 64], 0, hi);
      v16bf c01 = ldB_bf16(&k2T[buf][lo * 64], 32, hi);
      s0 = wmma_bf16(qe0, b00, s0);
      s0 = wmma_bf16(qe1, b01, s0);
      s0 = wmma_bf16(qr0, c00, s0);
      s0 = wmma_bf16(qr1, c01, s0);
      v16bf b10 = ldB_bf16(&kT[buf][(lo + 16) * 64], 0, hi);
      v16bf b11 = ldB_bf16(&kT[buf][(lo + 16) * 64], 32, hi);
      v16bf c10 = ldB_bf16(&k2T[buf][(lo + 16) * 64], 0, hi);
      v16bf c11 = ldB_bf16(&k2T[buf][(lo + 16) * 64], 32, hi);
      s1 = wmma_bf16(qe0, b10, s1);
      s1 = wmma_bf16(qe1, b11, s1);
      s1 = wmma_bf16(qr0, c10, s1);
      s1 = wmma_bf16(qr1, c11, s1);
    }
    bool msk0 = (kbase + lo) >= NTOK;        // pad-key masking (col fixed per lane)
    bool msk1 = (kbase + 16 + lo) >= NTOK;
    float p0[8], p1[8];
#pragma unroll
    for (int r = 0; r < 8; ++r) {
      float a = msk0 ? -1e30f : s0[r] * SCALEF;
      float c = msk1 ? -1e30f : s1[r] * SCALEF;
      float mx = fmaxf(a, c);
#pragma unroll
      for (int off = 1; off < 16; off <<= 1) mx = fmaxf(mx, __shfl_xor(mx, off));
      float mn = fmaxf(m[r], mx);
      float sc = __expf(m[r] - mn);
      float e0 = __expf(a - mn), e1 = __expf(c - mn);
      float rs = e0 + e1;
#pragma unroll
      for (int off = 1; off < 16; off <<= 1) rs += __shfl_xor(rs, off);
      l[r] = l[r] * sc + rs;
      m[r] = mn;
      p0[r] = e0;
      p1[r] = e1;
#pragma unroll
      for (int t = 0; t < 4; ++t) accO[t][r] *= sc;
    }
    // Re-layout P (C/D layout -> A layout) through per-wave LDS region.
#pragma unroll
    for (int r = 0; r < 8; ++r) {
      myP[(r + 8 * hi) * 32 + lo] = p0[r];
      myP[(r + 8 * hi) * 32 + 16 + lo] = p1[r];
    }
    wait_dscnt0();
    v16bf pa = ldA_f32(myP + lo * 32, 0, hi);  // P as A-operand (16x32)
#pragma unroll
    for (int t = 0; t < 4; ++t) {
      v16bf bv = ldB_bf16(&vT[buf][(t * 16 + lo) * 32], 0, hi);
      accO[t] = wmma_bf16(pa, bv, accO[t]);
    }
    wait_dscnt0();        // my LDS reads of this buf complete
    __syncthreads();      // safe to overwrite this buf next iteration
  }
  // Epilogue: divide by row sums, scatter to o[b2][tok][h2*64 + col] (matches
  // the reference's final transpose(1,0,2,3).reshape(b,n,AHD)).
  if (valid) {
#pragma unroll
    for (int t = 0; t < 4; ++t) {
#pragma unroll
      for (int r = 0; r < 8; ++r) {
        int tok = qtile * 16 + r + 8 * hi;
        if (tok < NTOK) {
          o[((size_t)b2 * NTOK + tok) * AHD_ + h2 * 64 + t * 16 + lo] =
              accO[t][r] * (1.0f / l[r]);
        }
      }
    }
  }
}

// ---------------- Head: LN(cls) @ head_w + head_b ----------------

__global__ void head_kernel(const float* __restrict__ x, const float* __restrict__ g,
                            const float* __restrict__ bvec, const float* __restrict__ hw,
                            const float* __restrict__ hb, float* __restrict__ out) {
  __shared__ float xs[D_];
  int b = blockIdx.x, lane = threadIdx.x;  // 32 threads
  const float* xr = x + (size_t)b * NTOK * D_;  // token 0 of batch b
  float v[4];
#pragma unroll
  for (int i = 0; i < 4; ++i) v[i] = xr[i * 32 + lane];
  float s = v[0] + v[1] + v[2] + v[3];
#pragma unroll
  for (int off = 1; off < 32; off <<= 1) s += __shfl_xor(s, off);
  float mean = s * (1.0f / 128.0f);
  float d2 = 0.f;
#pragma unroll
  for (int i = 0; i < 4; ++i) { float t = v[i] - mean; d2 += t * t; }
#pragma unroll
  for (int off = 1; off < 32; off <<= 1) d2 += __shfl_xor(d2, off);
  float inv = rsqrtf(d2 * (1.0f / 128.0f) + 1e-5f);
#pragma unroll
  for (int i = 0; i < 4; ++i) {
    int c = i * 32 + lane;
    xs[c] = (v[i] - mean) * inv * g[c] + bvec[c];
  }
  wait_dscnt0();
  if (lane < NC_) {
    float acc = hb[lane];
    for (int j = 0; j < D_; ++j) acc += xs[j] * hw[j * NC_ + lane];
    out[b * NC_ + lane] = acc;
  }
}

// ---------------- Host launcher ----------------

extern "C" void kernel_launch(void* const* d_in, const int* in_sizes, int n_in,
                              void* d_out, int out_size, void* d_ws, size_t ws_size,
                              hipStream_t stream) {
  (void)in_sizes; (void)n_in; (void)out_size; (void)ws_size;
  const float* x_eeg = (const float*)d_in[0];
  const float* est   = (const float*)d_in[1];
  const int*   cvi   = (const int*)d_in[2];
  const float* conv_w = (const float*)d_in[3];
  const float* conv_b = (const float*)d_in[4];
  const float* cls_token = (const float*)d_in[5];
  const float* cls_pos   = (const float*)d_in[6];
  const float* bte = (const float*)d_in[7];
  const float* btr = (const float*)d_in[8];
  const float* bcr = (const float*)d_in[9];
  const float* bce = (const float*)d_in[10];
  const float* ln1_g = (const float*)d_in[11];
  const float* ln1_b = (const float*)d_in[12];
  const float* wqkv = (const float*)d_in[13];
  const float* wkt  = (const float*)d_in[14];
  const float* wkc  = (const float*)d_in[15];
  const float* wo   = (const float*)d_in[16];
  const float* bo   = (const float*)d_in[17];
  const float* ln2_g = (const float*)d_in[18];
  const float* ln2_b = (const float*)d_in[19];
  const float* w1 = (const float*)d_in[20];
  const float* b1 = (const float*)d_in[21];
  const float* w2 = (const float*)d_in[22];
  const float* b2 = (const float*)d_in[23];
  const float* hln_g = (const float*)d_in[24];
  const float* hln_b = (const float*)d_in[25];
  const float* head_w = (const float*)d_in[26];
  const float* head_b = (const float*)d_in[27];

  size_t off = 0;
  auto alloc = [&](size_t bytes) -> char* {
    char* p = (char*)d_ws + off;
    off = (off + bytes + 255) & ~(size_t)255;
    return p;
  };
  float* xb   = (float*)alloc((size_t)ROWS * D_ * 4);
  float* rtb  = (float*)alloc((size_t)ROWS * D_ * 4);
  float* rcb  = (float*)alloc((size_t)ROWS * D_ * 4);
  float* xnb  = (float*)alloc((size_t)ROWS * D_ * 4);
  float* qkvb = (float*)alloc((size_t)ROWS * 1536 * 4);
  float* ktb  = (float*)alloc((size_t)ROWS * 512 * 4);
  float* kcb  = (float*)alloc((size_t)ROWS * 512 * 4);
  float* ob   = (float*)alloc((size_t)ROWS * 512 * 4);
  float* hdnb = (float*)alloc((size_t)ROWS * 512 * 4);
  __bf16* qeb = (__bf16*)alloc((size_t)GRP * NPAD * 64 * 2);
  __bf16* qrb = (__bf16*)alloc((size_t)GRP * NPAD * 64 * 2);
  __bf16* kbb = (__bf16*)alloc((size_t)GRP * NPAD * 64 * 2);
  __bf16* k2b = (__bf16*)alloc((size_t)GRP * NPAD * 64 * 2);
  __bf16* vtb = (__bf16*)alloc((size_t)GRP * 64 * NPAD * 2);
  __bf16* wqkvT = (__bf16*)alloc((size_t)1536 * 128 * 2);
  __bf16* wktT  = (__bf16*)alloc((size_t)512 * 128 * 2);
  __bf16* wkcT  = (__bf16*)alloc((size_t)512 * 128 * 2);
  __bf16* woT   = (__bf16*)alloc((size_t)128 * 512 * 2);
  __bf16* w1T   = (__bf16*)alloc((size_t)512 * 128 * 2);
  __bf16* w2T   = (__bf16*)alloc((size_t)128 * 512 * 2);

  auto gemm = [&](const float* A, const __bf16* Wt, const float* bias,
                  const float* resid, float* Cb, int M, int K, int N, int flags) {
    int tm = (M + 15) / 16, nt = N / 64, tot = tm * nt;
    gemm_wmma<<<(tot + 7) / 8, 256, 0, stream>>>(A, Wt, bias, resid, Cb, M, K, N,
                                                 tot, nt, flags);
  };
  auto transp = [&](const float* in, __bf16* out, int K, int N) {
    transpose_cast<<<(K * N + 255) / 256, 256, 0, stream>>>(in, out, K, N);
  };

  embed_kernel<<<ROWS, 128, 0, stream>>>(x_eeg, est, cvi, conv_w, conv_b, cls_token,
                                         cls_pos, xb, rtb, rcb);

  for (int l = 0; l < L_; ++l) {
    transp(wqkv + (size_t)l * 128 * 1536, wqkvT, 128, 1536);
    transp(wkt + (size_t)l * 128 * 512, wktT, 128, 512);
    transp(wkc + (size_t)l * 128 * 512, wkcT, 128, 512);
    transp(wo + (size_t)l * 512 * 128, woT, 512, 128);
    transp(w1 + (size_t)l * 128 * 512, w1T, 128, 512);
    transp(w2 + (size_t)l * 512 * 128, w2T, 512, 128);

    layernorm_k<<<(ROWS + 7) / 8, 256, 0, stream>>>(xb, ln1_g + l * 128,
                                                    ln1_b + l * 128, xnb, ROWS);
    gemm(xnb, wqkvT, nullptr, nullptr, qkvb, ROWS, 128, 1536, 0);
    gemm(rtb, wktT, nullptr, nullptr, ktb, ROWS, 128, 512, 0);
    gemm(rcb, wkcT, nullptr, nullptr, kcb, ROWS, 128, 512, 0);

    attn_prep_k<<<dim3(NPAD, GRP), 64, 0, stream>>>(qkvb, ktb, kcb, bte, btr, bcr,
                                                    bce, qeb, qrb, kbb, k2b, vtb);
    flash_attn_k<<<dim3((QT + WPB - 1) / WPB, GRP), WPB * 32, 0, stream>>>(
        qeb, qrb, kbb, k2b, vtb, ob);

    gemm(ob, woT, bo + l * 128, xb, xb, ROWS, 512, 128, 1 | 2);
    layernorm_k<<<(ROWS + 7) / 8, 256, 0, stream>>>(xb, ln2_g + l * 128,
                                                    ln2_b + l * 128, xnb, ROWS);
    gemm(xnb, w1T, b1 + l * 512, nullptr, hdnb, ROWS, 128, 512, 1 | 4);
    gemm(hdnb, w2T, b2 + l * 128, xb, xb, ROWS, 512, 128, 1 | 2);
  }

  head_kernel<<<B_, 32, 0, stream>>>(xb, hln_g, hln_b, head_w, head_b, (float*)d_out);
}
